// SelfAttention2D_81226421502392
// MI455X (gfx1250) — matrix operand order repaired and checked by reading, not compiled, not run on hardware
//
#include <hip/hip_runtime.h>

// ---------------------------------------------------------------------------
// SelfAttention2D on MI455X (gfx1250): all GEMM stages via v_wmma_f32_16x16x32_bf16
// B=8, C=256, H=W=128.  wave32, 256-thread blocks = 8 waves, 16x16 tile/wave.
// ---------------------------------------------------------------------------

typedef __bf16 bf16_t;
typedef __attribute__((ext_vector_type(16))) __bf16 v16bf;
typedef __attribute__((ext_vector_type(8)))  float  v8f;

#define CCH 256
#define BB  8
#define HH  128
#define WW  128
#define HWP (HH * WW)          // 16384 pixels per (b, channel-plane)

// ---- WMMA fragment loaders (contiguous k order: k = k0 + 16*hi + e in both A and B) -------

// A-matrix 16x32 bf16. Generic element strides: A[r,k] at base[r*rs + k*cs].
// lane<16:  row = lane,    elems 0..15 -> K = k0+0..15
// lane>=16: row = lane-16, elems 0..15 -> K = k0+16..31
__device__ __forceinline__ v16bf load_frag_a_bf16(const bf16_t* A, int r0, int k0,
                                                  long rs, long cs) {
  const int lane = threadIdx.x & 31;
  const int l = lane & 15, hi = lane >> 4;
  const long rbase = (long)(r0 + l) * rs;
  v16bf a;
#pragma unroll
  for (int e = 0; e < 8; ++e) {
    a[e]     = A[rbase + (long)(k0 + hi * 16 + e) * cs];
    a[e + 8] = A[rbase + (long)(k0 + hi * 16 + 8 + e) * cs];
  }
  return a;
}

// B-matrix 32x16 bf16. B[k,n] at base[k*rs + n*cs].
// lane&15 = column n; lane<16 covers K rows k0+0..15, lane>=16 covers k0+16..31;
// VGPR j holds rows (2j, 2j+1) of its half in [15:0],[31:16].
__device__ __forceinline__ v16bf load_frag_b_bf16(const bf16_t* Bm, int k0, int c0,
                                                  long rs, long cs) {
  const int lane = threadIdx.x & 31;
  const int l = lane & 15, hi = lane >> 4;
  const long cbase = (long)(c0 + l) * cs;
  v16bf b;
#pragma unroll
  for (int j = 0; j < 8; ++j) {
    b[2 * j]     = Bm[(long)(k0 + hi * 16 + 2 * j) * rs + cbase];
    b[2 * j + 1] = Bm[(long)(k0 + hi * 16 + 2 * j + 1) * rs + cbase];
  }
  return b;
}

// Same B layout but gathering from an fp32 source, converting to bf16 in-register.
__device__ __forceinline__ v16bf load_frag_b_f32src(const float* Bm, int k0, int c0,
                                                    long rs, long cs) {
  const int lane = threadIdx.x & 31;
  const int l = lane & 15, hi = lane >> 4;
  const long cbase = (long)(c0 + l) * cs;
  v16bf b;
#pragma unroll
  for (int j = 0; j < 8; ++j) {
    b[2 * j]     = (bf16_t)Bm[(long)(k0 + hi * 16 + 2 * j) * rs + cbase];
    b[2 * j + 1] = (bf16_t)Bm[(long)(k0 + hi * 16 + 2 * j + 1) * rs + cbase];
  }
  return b;
}

// C/D 16x16 f32: VGPR i -> row r0 + hi*8 + i, col = c0 + (lane&15).
__device__ __forceinline__ void store_frag_f32(float* Cm, const v8f c, int r0, int c0,
                                               long rs) {
  const int lane = threadIdx.x & 31;
  const int l = lane & 15, hi = lane >> 4;
#pragma unroll
  for (int i = 0; i < 8; ++i)
    Cm[(long)(r0 + hi * 8 + i) * rs + (c0 + l)] = c[i];
}

__device__ __forceinline__ void store_frag_bf16(bf16_t* Cm, const v8f c, int r0, int c0,
                                                long rs) {
  const int lane = threadIdx.x & 31;
  const int l = lane & 15, hi = lane >> 4;
#pragma unroll
  for (int i = 0; i < 8; ++i)
    Cm[(long)(r0 + hi * 8 + i) * rs + (c0 + l)] = (bf16_t)c[i];
}

__device__ __forceinline__ void store_frag_bias_bf16(bf16_t* Cm, const v8f c,
                                                     const float* bias, int r0, int c0,
                                                     long rs) {
  const int lane = threadIdx.x & 31;
  const int l = lane & 15, hi = lane >> 4;
#pragma unroll
  for (int i = 0; i < 8; ++i) {
    const int r = r0 + hi * 8 + i;
    Cm[(long)r * rs + (c0 + l)] = (bf16_t)(c[i] + bias[r]);
  }
}

__device__ __forceinline__ v8f wmma_bf16(v16bf a, v16bf b, v8f c) {
  return __builtin_amdgcn_wmma_f32_16x16x32_bf16(false, a, false, b, (short)0, c,
                                                 false, false);
}

// ---- Kernel 0: fp32 -> bf16 weight conversion -------------------------------
__global__ void cvt_weights(const float* wk, const float* wq, const float* wv,
                            const float* wo, bf16_t* wkb, bf16_t* wqb, bf16_t* wvb,
                            bf16_t* wob) {
  const int i = blockIdx.x * blockDim.x + threadIdx.x;
  if (i < CCH * CCH) {
    wkb[i] = (bf16_t)wk[i];
    wqb[i] = (bf16_t)wq[i];
    wvb[i] = (bf16_t)wv[i];
    wob[i] = (bf16_t)wo[i];
  }
}

// ---- Kernel 1: fused k/q/v 1x1 conv -----------------------------------------
// out_t[b,c,p] = sum_i w_t[c,i] * x[b,i,p] + b_t[c].  GEMM [256x256]x[256,16384]/batch.
// grid: x = HWP/32 pixel tiles, y = C/64 row tiles, z = batch. 8 waves -> 4x2 tiles.
__global__ void conv_kqv(const float* __restrict__ x, const bf16_t* __restrict__ wkb,
                         const bf16_t* __restrict__ wqb, const bf16_t* __restrict__ wvb,
                         const float* __restrict__ bk, const float* __restrict__ bq,
                         const float* __restrict__ bv, bf16_t* __restrict__ kout,
                         bf16_t* __restrict__ qout, bf16_t* __restrict__ vout) {
  const int b = blockIdx.z;
  const int wid = threadIdx.x >> 5;
  const int row0 = blockIdx.y * 64 + (wid & 3) * 16;
  const int col0 = blockIdx.x * 32 + (wid >> 2) * 16;
  const float* xb = x + (long)b * CCH * HWP;

  v8f ak = {}, aq = {}, av = {};
#pragma unroll
  for (int k0 = 0; k0 < CCH; k0 += 32) {
    const v16bf fb = load_frag_b_f32src(xb, k0, col0, HWP, 1);
    ak = wmma_bf16(load_frag_a_bf16(wkb, row0, k0, CCH, 1), fb, ak);
    aq = wmma_bf16(load_frag_a_bf16(wqb, row0, k0, CCH, 1), fb, aq);
    av = wmma_bf16(load_frag_a_bf16(wvb, row0, k0, CCH, 1), fb, av);
  }
  const long ob = (long)b * CCH * HWP;
  store_frag_bias_bf16(kout + ob, ak, bk, row0, col0, HWP);
  store_frag_bias_bf16(qout + ob, aq, bq, row0, col0, HWP);
  store_frag_bias_bf16(vout + ob, av, bv, row0, col0, HWP);
}

// ---- Kernel 2: scores[b,c] = K^T @ Q  (128x128, K-dim = H = 128) ------------
// A[r,k] = kmat[k*W + r]  (transpose via strides: rs=1, cs=W).
// grid: x = W/32, y = W/64, z = b*C + c.
__global__ void scores_gemm(const bf16_t* __restrict__ kbuf,
                            const bf16_t* __restrict__ qbuf,
                            float* __restrict__ scores) {
  const int bc = blockIdx.z;
  const bf16_t* km = kbuf + (long)bc * HWP;
  const bf16_t* qm = qbuf + (long)bc * HWP;
  float* sm = scores + (long)bc * WW * WW;
  const int wid = threadIdx.x >> 5;
  const int row0 = blockIdx.y * 64 + (wid & 3) * 16;
  const int col0 = blockIdx.x * 32 + (wid >> 2) * 16;

  v8f acc = {};
#pragma unroll
  for (int k0 = 0; k0 < HH; k0 += 32) {
    acc = wmma_bf16(load_frag_a_bf16(km, row0, k0, /*rs=*/1, /*cs=*/WW),
                    load_frag_b_bf16(qm, k0, col0, /*rs=*/WW, /*cs=*/1), acc);
  }
  store_frag_f32(sm, acc, row0, col0, WW);
}

// ---- Kernel 3: softmax over batch axis (8 values / site) --------------------
__global__ void softmax_batch(const float* __restrict__ scores,
                              bf16_t* __restrict__ attn) {
  const long n = (long)CCH * WW * WW;
  const long idx = (long)blockIdx.x * blockDim.x + threadIdx.x;
  if (idx >= n) return;
  float v[BB];
  float m = -3.402823466e+38f;
#pragma unroll
  for (int b = 0; b < BB; ++b) {
    v[b] = scores[(long)b * n + idx];
    m = fmaxf(m, v[b]);
  }
  float s = 0.f;
#pragma unroll
  for (int b = 0; b < BB; ++b) {
    v[b] = __expf(v[b] - m);
    s += v[b];
  }
  const float inv = 1.f / s;
#pragma unroll
  for (int b = 0; b < BB; ++b) attn[(long)b * n + idx] = (bf16_t)(v[b] * inv);
}

// ---- Kernel 4: out2[b,c] = V @ attn  (128x128, K-dim = W = 128) -------------
__global__ void out_gemm(const bf16_t* __restrict__ vbuf,
                         const bf16_t* __restrict__ attn,
                         bf16_t* __restrict__ out2) {
  const int bc = blockIdx.z;
  const bf16_t* vm = vbuf + (long)bc * HWP;
  const bf16_t* am = attn + (long)bc * WW * WW;
  bf16_t* om = out2 + (long)bc * HWP;
  const int wid = threadIdx.x >> 5;
  const int row0 = blockIdx.y * 64 + (wid & 3) * 16;
  const int col0 = blockIdx.x * 32 + (wid >> 2) * 16;

  v8f acc = {};
#pragma unroll
  for (int k0 = 0; k0 < WW; k0 += 32) {
    acc = wmma_bf16(load_frag_a_bf16(vm, row0, k0, /*rs=*/WW, /*cs=*/1),
                    load_frag_b_bf16(am, k0, col0, /*rs=*/WW, /*cs=*/1), acc);
  }
  store_frag_bf16(om, acc, row0, col0, WW);
}

// ---- Kernel 5: output 1x1 conv + bias + residual, fp32 out ------------------
__global__ void conv_o_residual(const bf16_t* __restrict__ out2,
                                const bf16_t* __restrict__ wob,
                                const float* __restrict__ bo,
                                const float* __restrict__ x, float* __restrict__ out) {
  const int b = blockIdx.z;
  const bf16_t* src = out2 + (long)b * CCH * HWP;
  const float* xb = x + (long)b * CCH * HWP;
  float* ob = out + (long)b * CCH * HWP;
  const int wid = threadIdx.x >> 5;
  const int row0 = blockIdx.y * 64 + (wid & 3) * 16;
  const int col0 = blockIdx.x * 32 + (wid >> 2) * 16;

  v8f acc = {};
#pragma unroll
  for (int k0 = 0; k0 < CCH; k0 += 32) {
    acc = wmma_bf16(load_frag_a_bf16(wob, row0, k0, CCH, 1),
                    load_frag_b_bf16(src, k0, col0, HWP, 1), acc);
  }
  const int lane = threadIdx.x & 31;
  const int l = lane & 15, hi = lane >> 4;
#pragma unroll
  for (int i = 0; i < 8; ++i) {
    const int r = row0 + hi * 8 + i;
    const long off = (long)r * HWP + (col0 + l);
    ob[off] = acc[i] + bo[r] + xb[off];
  }
}

// ---------------------------------------------------------------------------
extern "C" void kernel_launch(void* const* d_in, const int* in_sizes, int n_in,
                              void* d_out, int out_size, void* d_ws, size_t ws_size,
                              hipStream_t stream) {
  const float* x  = (const float*)d_in[0];
  const float* wk = (const float*)d_in[1];
  const float* bk = (const float*)d_in[2];
  const float* wq = (const float*)d_in[3];
  const float* bq = (const float*)d_in[4];
  const float* wv = (const float*)d_in[5];
  const float* bv = (const float*)d_in[6];
  const float* wo = (const float*)d_in[7];
  const float* bo = (const float*)d_in[8];
  float* out = (float*)d_out;

  // Workspace layout (bf16 elements). bufA: k -> reused as out2; bufB: q -> attn;
  // bufC: v.  fp32 scores staged in d_out (dead before final conv rewrites it).
  const long NTENS = (long)BB * CCH * HWP;  // 33,554,432
  bf16_t* wkb  = (bf16_t*)d_ws;
  bf16_t* wqb  = wkb + CCH * CCH;
  bf16_t* wvb  = wqb + CCH * CCH;
  bf16_t* wob  = wvb + CCH * CCH;
  bf16_t* bufA = wob + CCH * CCH;  // k, later out2
  bf16_t* bufB = bufA + NTENS;     // q, later attn
  bf16_t* bufC = bufB + NTENS;     // v
  float* scores = (float*)d_out;

  // 0) weights -> bf16
  cvt_weights<<<(CCH * CCH + 255) / 256, 256, 0, stream>>>(wk, wq, wv, wo, wkb, wqb,
                                                           wvb, wob);
  // 1) fused k/q/v conv (reads fp32 x once, converts in-register)
  conv_kqv<<<dim3(HWP / 32, CCH / 64, BB), 256, 0, stream>>>(x, wkb, wqb, wvb, bk, bq,
                                                             bv, bufA, bufB, bufC);
  // 2) scores = K^T Q per (b,c), fp32 into d_out scratch
  scores_gemm<<<dim3(WW / 32, WW / 64, BB * CCH), 256, 0, stream>>>(bufA, bufB, scores);
  // 3) softmax over batch axis -> bf16 attn (reuses q buffer)
  {
    const long n = (long)CCH * WW * WW;
    softmax_batch<<<(unsigned)((n + 255) / 256), 256, 0, stream>>>(scores, bufB);
  }
  // 4) out2 = V @ attn per (b,c) -> bf16 (reuses k buffer)
  out_gemm<<<dim3(WW / 32, HH / 64, BB * CCH), 256, 0, stream>>>(bufC, bufB, bufA);
  // 5) output conv + bias + residual, fp32 -> d_out (overwrites scores scratch)
  conv_o_residual<<<dim3(HWP / 32, CCH / 64, BB), 256, 0, stream>>>(bufA, wob, bo, x,
                                                                    out);
}
